// SpectralMoEDictionaryCrossAttention_8864812499093
// MI455X (gfx1250) — compile-verified, matrix operand order checked by reading, not run on hardware
//
#include <hip/hip_runtime.h>
#include <math.h>

// ============================================================================
// SpectralMoEDictionaryCrossAttention forward for MI455X (gfx1250, wave32).
// All dense projections run through a single WMMA (v_wmma_f32_16x16x32_f16)
// tiled GEMM with f32 accumulate; everything else is fused elementwise work.
//
// GEMM LDS tiles are stored pre-swizzled into the CDNA5 16-bit fragment
// order (K blocks [0..7 | 16..23 | 8..15 | 24..31]) so each lane's 16-half
// fragment is a contiguous 32-byte run -> two ds_load_b128, no scalar packs.
//
// Param order assumption: d_in follows setup_inputs() dict insertion order,
// nested dicts expanded in insertion order (x, then _make_params() literal
// order). See index constants below.
// ============================================================================

typedef __attribute__((ext_vector_type(16))) _Float16 v16h;
typedef __attribute__((ext_vector_type(8)))  _Float16 v8h;
typedef __attribute__((ext_vector_type(4)))  _Float16 v4h;
typedef __attribute__((ext_vector_type(8)))  float    v8f;
typedef __attribute__((ext_vector_type(4)))  float    v4f;

#define DEV __device__ __forceinline__

DEV float gelu_f(float x) { return 0.5f * x * (1.0f + erff(x * 0.70710678118654752f)); }
DEV float sigmoid_f(float x) { return 1.0f / (1.0f + expf(-x)); }

// permuted column position for K within a 32-wide tile:
// blocks [0..7]->0, [8..15]->2, [16..23]->1, [24..31]->3
DEV int kpos(int k) {
    int blk = k >> 3;
    int pb = (blk == 1) ? 2 : (blk == 2) ? 1 : blk;
    return pb * 8 + (k & 7);
}

// ---------------------------------------------------------------------------
// Generic GEMM: Co[M,N] = epi(alpha * A[M,K] @ B + bias)
//   B element (k,n) = transB ? Wm[n*K+k] : Wm[k*N+n]
//   epi: 0=none, 1=gelu.  A row stride lda, C row stride ldc.
// Tile: 64x64, K-step 32, 4 waves (128 thr).  K must be a multiple of 32 and
// N a multiple of 4 (true for every call site here).
// ---------------------------------------------------------------------------
__global__ void __launch_bounds__(128)
k_gemm(const float* __restrict__ A, const float* __restrict__ Wm,
       const float* __restrict__ bias, float* __restrict__ Co,
       int M, int N, int K, int lda, int ldc, int transB, int epi, float alpha)
{
    // [row][permuted-k], stride 40 halves = 80B: 16B-aligned rows, and the 16
    // fragment rows land on 16 distinct banks (20*r mod 64 is a permutation).
    __shared__ __align__(16) _Float16 As[64][40];
    __shared__ __align__(16) _Float16 Bs[64][40];   // [n][permuted-k]

    const int tid  = threadIdx.x;
    const int lane = tid & 31;
    const int wave = tid >> 5;
    const int m0   = blockIdx.y * 64;
    const int n0   = blockIdx.x * 64;
    const int hh   = lane >> 4;    // half-wave select
    const int rl   = lane & 15;    // row/col within 16

    v8f acc[4] = {v8f{}, v8f{}, v8f{}, v8f{}};

    for (int k0 = 0; k0 < K; k0 += 32) {
        // ---- stage A tile 64x32: float4 global loads, v4h LDS stores ----
        for (int p = tid; p < 64 * 8; p += 128) {
            int r  = p >> 3;
            int kq = (p & 7) * 4;           // 0,4,...,28
            int gm = m0 + r;
            v4f v = v4f{0.f, 0.f, 0.f, 0.f};
            if (gm < M)
                v = *(const v4f*)(A + (size_t)gm * lda + (k0 + kq));
            v4h h = {(_Float16)v.x, (_Float16)v.y, (_Float16)v.z, (_Float16)v.w};
            *(v4h*)&As[r][kpos(kq)] = h;
        }
        // ---- stage B tile (as [n][k]) ----
        if (transB) {
            for (int p = tid; p < 64 * 8; p += 128) {
                int n  = p >> 3;
                int kq = (p & 7) * 4;
                int gn = n0 + n;
                v4f v = v4f{0.f, 0.f, 0.f, 0.f};
                if (gn < N)
                    v = *(const v4f*)(Wm + (size_t)gn * K + (k0 + kq));
                v4h h = {(_Float16)v.x, (_Float16)v.y, (_Float16)v.z, (_Float16)v.w};
                *(v4h*)&Bs[n][kpos(kq)] = h;
            }
        } else {
            for (int p = tid; p < 32 * 16; p += 128) {
                int k  = p >> 4;
                int n4 = (p & 15) * 4;
                int gn = n0 + n4;
                v4f v = v4f{0.f, 0.f, 0.f, 0.f};
                if (gn < N)                  // N multiple of 4 -> whole quad ok
                    v = *(const v4f*)(Wm + (size_t)(k0 + k) * N + gn);
                int pos = kpos(k);
                Bs[n4 + 0][pos] = (_Float16)v.x;
                Bs[n4 + 1][pos] = (_Float16)v.y;
                Bs[n4 + 2][pos] = (_Float16)v.z;
                Bs[n4 + 3][pos] = (_Float16)v.w;
            }
        }
        __syncthreads();

        // ---- fragments: contiguous 32B per lane -> 2x ds_load_b128 ----
        const _Float16* arow = &As[wave * 16 + rl][hh * 16];
        v8h a0 = *(const v8h*)arow;
        v8h a1 = *(const v8h*)(arow + 8);
        v16h af = __builtin_shufflevector(a0, a1, 0, 1, 2, 3, 4, 5, 6, 7,
                                          8, 9, 10, 11, 12, 13, 14, 15);
#pragma unroll
        for (int nt = 0; nt < 4; ++nt) {
            const _Float16* brow = &Bs[nt * 16 + rl][hh * 16];
            v8h b0 = *(const v8h*)brow;
            v8h b1 = *(const v8h*)(brow + 8);
            v16h bf = __builtin_shufflevector(b0, b1, 0, 1, 2, 3, 4, 5, 6, 7,
                                              8, 9, 10, 11, 12, 13, 14, 15);
            acc[nt] = __builtin_amdgcn_wmma_f32_16x16x32_f16(
                false, af, false, bf, (short)0, acc[nt], false, false);
        }
        __syncthreads();
    }

    // epilogue: C/D layout -> VGPR r holds M = r + 8*hh, N = rl (+16*nt)
#pragma unroll
    for (int nt = 0; nt < 4; ++nt) {
#pragma unroll
        for (int r = 0; r < 8; ++r) {
            int gm = m0 + wave * 16 + r + 8 * hh;
            int gn = n0 + nt * 16 + rl;
            if (gm < M && gn < N) {
                float v = acc[nt][r] * alpha;
                if (bias) v += bias[gn];
                if (epi == 1) v = gelu_f(v);
                Co[(size_t)gm * ldc + gn] = v;
            }
        }
    }
}

// ---------------------------------------------------------------------------
// Depthwise 3x3, pad 1, NHWC.  Input may be a channel-slice of a wider buffer
// (ldx / offx).  Optional gelu on input (pre) or output (post).
// ---------------------------------------------------------------------------
__global__ void k_dwconv3x3(const float* __restrict__ x, const float* __restrict__ w,
                            const float* __restrict__ b, float* __restrict__ y,
                            int Bn, int Hn, int Wn, int Cn, int ldx, int offx,
                            int pre_gelu, int post_gelu)
{
    size_t idx = (size_t)blockIdx.x * blockDim.x + threadIdx.x;
    size_t total = (size_t)Bn * Hn * Wn * Cn;
    if (idx >= total) return;
    int c = (int)(idx % Cn);
    size_t p = idx / Cn;
    int wq = (int)(p % Wn); p /= Wn;
    int hq = (int)(p % Hn); p /= Hn;
    int bb = (int)p;
    float s = b ? b[c] : 0.0f;
#pragma unroll
    for (int dy = 0; dy < 3; ++dy) {
        int hy = hq + dy - 1;
        if (hy < 0 || hy >= Hn) continue;
#pragma unroll
        for (int dx = 0; dx < 3; ++dx) {
            int wx = wq + dx - 1;
            if (wx < 0 || wx >= Wn) continue;
            float xv = x[(((size_t)bb * Hn + hy) * Wn + wx) * ldx + offx + c];
            if (pre_gelu) xv = gelu_f(xv);
            s += xv * w[c * 9 + dy * 3 + dx];
        }
    }
    if (post_gelu) s = gelu_f(s);
    y[idx] = s;
}

// o = a + beta*b (elementwise over n floats)
__global__ void k_addsub(const float* __restrict__ a, const float* __restrict__ b,
                         float* __restrict__ o, size_t n, float beta)
{
    size_t i = (size_t)blockIdx.x * blockDim.x + threadIdx.x;
    if (i < n) o[i] = a[i] + beta * b[i];
}

// channel-strided copy: dst[r*ldd+offd+c] = src[r*lds+offs+c]
__global__ void k_copych(const float* __restrict__ src, float* __restrict__ dst,
                         size_t rows, int Cn, int lds, int offs, int ldd, int offd)
{
    size_t idx = (size_t)blockIdx.x * blockDim.x + threadIdx.x;
    if (idx >= rows * (size_t)Cn) return;
    int c = (int)(idx % Cn);
    size_t r = idx / Cn;
    dst[r * ldd + offd + c] = src[r * lds + offs + c];
}

// parity gather along H (axis=2) or W (axis=3), NHWC
__global__ void k_slice(const float* __restrict__ src, float* __restrict__ dst,
                        int Bn, int Ho, int Wo, int Cn, int Hi, int Wi, int axis, int par)
{
    size_t idx = (size_t)blockIdx.x * blockDim.x + threadIdx.x;
    size_t total = (size_t)Bn * Ho * Wo * Cn;
    if (idx >= total) return;
    int c = (int)(idx % Cn);
    size_t p = idx / Cn;
    int w = (int)(p % Wo); p /= Wo;
    int h = (int)(p % Ho); p /= Ho;
    int b = (int)p;
    int hs = (axis == 2) ? 2 * h + par : h;
    int ws = (axis == 3) ? 2 * w + par : w;
    dst[idx] = src[(((size_t)b * Hi + hs) * Wi + ws) * Cn + c];
}

// interleave even/odd along H (axis=2) or W (axis=3); dst dims already doubled
__global__ void k_interleave(const float* __restrict__ ev, const float* __restrict__ od,
                             float* __restrict__ dst, int Bn, int Ho, int Wo, int Cn, int axis)
{
    size_t idx = (size_t)blockIdx.x * blockDim.x + threadIdx.x;
    size_t total = (size_t)Bn * Ho * Wo * Cn;
    if (idx >= total) return;
    int c = (int)(idx % Cn);
    size_t p = idx / Cn;
    int w = (int)(p % Wo); p /= Wo;
    int h = (int)(p % Ho); p /= Ho;
    int b = (int)p;
    const float* s;
    int hs = h, ws = w;
    int Hi = Ho, Wi = Wo;
    if (axis == 2) { s = (h & 1) ? od : ev; hs = h >> 1; Hi = Ho >> 1; }
    else           { s = (w & 1) ? od : ev; ws = w >> 1; Wi = Wo >> 1; }
    dst[idx] = s[(((size_t)b * Hi + hs) * Wi + ws) * Cn + c];
}

// layernorm over last dim C; one 128-thread block per row
__global__ void __launch_bounds__(128)
k_layernorm(const float* __restrict__ x, const float* __restrict__ g,
            const float* __restrict__ b, float* __restrict__ y, int C)
{
    __shared__ float r1[128];
    __shared__ float r2[128];
    size_t row = blockIdx.x;
    const float* xr = x + row * C;
    float s = 0.f, s2 = 0.f;
    for (int c = threadIdx.x; c < C; c += 128) { float v = xr[c]; s += v; s2 += v * v; }
    r1[threadIdx.x] = s; r2[threadIdx.x] = s2;
    __syncthreads();
    for (int off = 64; off > 0; off >>= 1) {
        if (threadIdx.x < off) { r1[threadIdx.x] += r1[threadIdx.x + off];
                                 r2[threadIdx.x] += r2[threadIdx.x + off]; }
        __syncthreads();
    }
    float m = r1[0] / C;
    float var = r2[0] / C - m * m;
    float inv = rsqrtf(var + 1e-5f);
    for (int c = threadIdx.x; c < C; c += 128)
        y[row * C + c] = (xr[c] - m) * inv * g[c] + b[c];
}

// in-place softmax over last dim L; one 64-thread block per row
__global__ void __launch_bounds__(64)
k_softmax(float* __restrict__ x, int L)
{
    __shared__ float red[64];
    size_t row = blockIdx.x;
    float* xr = x + row * L;
    float mx = -3.4e38f;
    for (int i = threadIdx.x; i < L; i += 64) mx = fmaxf(mx, xr[i]);
    red[threadIdx.x] = mx; __syncthreads();
    for (int off = 32; off > 0; off >>= 1) {
        if (threadIdx.x < off) red[threadIdx.x] = fmaxf(red[threadIdx.x], red[threadIdx.x + off]);
        __syncthreads();
    }
    mx = red[0]; __syncthreads();
    float s = 0.f;
    for (int i = threadIdx.x; i < L; i += 64) { float e = expf(xr[i] - mx); xr[i] = e; s += e; }
    red[threadIdx.x] = s; __syncthreads();
    for (int off = 32; off > 0; off >>= 1) {
        if (threadIdx.x < off) red[threadIdx.x] += red[threadIdx.x + off];
        __syncthreads();
    }
    float inv = 1.0f / red[0];
    for (int i = threadIdx.x; i < L; i += 64) xr[i] *= inv;
}

// top-2 router: softmax probs, top2 on logits+bias, renormalized weights
__global__ void k_router(const float* __restrict__ lg, const float* __restrict__ eb,
                         float* __restrict__ rw, int rows)
{
    int r = blockIdx.x * blockDim.x + threadIdx.x;
    if (r >= rows) return;
    float l[4], t[4], p[4];
#pragma unroll
    for (int c = 0; c < 4; ++c) l[c] = lg[(size_t)r * 4 + c];
    float mx = fmaxf(fmaxf(l[0], l[1]), fmaxf(l[2], l[3]));
    float s = 0.f;
#pragma unroll
    for (int c = 0; c < 4; ++c) { p[c] = expf(l[c] - mx); s += p[c]; }
#pragma unroll
    for (int c = 0; c < 4; ++c) { p[c] /= s; t[c] = l[c] + eb[c]; }
    int i1 = 0;
#pragma unroll
    for (int c = 1; c < 4; ++c) if (t[c] > t[i1]) i1 = c;
    int i2 = -1;
#pragma unroll
    for (int c = 0; c < 4; ++c)
        if (c != i1 && (i2 < 0 || t[c] > t[i2])) i2 = c;
    float sum2 = p[i1] + p[i2] + 1e-8f;
#pragma unroll
    for (int c = 0; c < 4; ++c)
        rw[(size_t)r * 4 + c] = (c == i1 || c == i2) ? p[c] / sum2 : 0.f;
}

// out[r,c] += rw[r*4+k] * eo[r,c]   (C=384)
__global__ void k_combine(const float* __restrict__ eo, const float* __restrict__ rw,
                          float* __restrict__ outp, size_t rows, int k)
{
    size_t idx = (size_t)blockIdx.x * blockDim.x + threadIdx.x;
    if (idx >= rows * 384) return;
    size_t r = idx / 384;
    outp[idx] += rw[r * 4 + k] * eo[idx];
}

// dst[r,c] += rs[c] * src[r,c] * (sa ? sa[r] : 1)
__global__ void k_scale_res(float* __restrict__ dst, const float* __restrict__ src,
                            const float* __restrict__ rs, const float* __restrict__ sa,
                            size_t rows, int Cn)
{
    size_t idx = (size_t)blockIdx.x * blockDim.x + threadIdx.x;
    if (idx >= rows * (size_t)Cn) return;
    int c = (int)(idx % Cn);
    size_t r = idx / Cn;
    float v = src[idx];
    if (sa) v *= sa[r];
    dst[idx] += rs[c] * v;
}

// g[r,c] = gelu(a[r,c]) * f[r*ldf + offf + c]
__global__ void k_gelu_mul(const float* __restrict__ a, const float* __restrict__ f,
                           float* __restrict__ g, size_t rows, int Cn, int ldf, int offf)
{
    size_t idx = (size_t)blockIdx.x * blockDim.x + threadIdx.x;
    if (idx >= rows * (size_t)Cn) return;
    int c = (int)(idx % Cn);
    size_t r = idx / Cn;
    g[idx] = gelu_f(a[idx]) * f[r * ldf + offf + c];
}

// per-pixel channel mean & max -> mm[0..rows) mean, mm[rows..2rows) max
__global__ void k_poolmm(const float* __restrict__ s, float* __restrict__ mm,
                         size_t rows, int C)
{
    size_t r = (size_t)blockIdx.x * blockDim.x + threadIdx.x;
    if (r >= rows) return;
    const float* xr = s + r * C;
    float acc = 0.f, mx = -3.4e38f;
    for (int c = 0; c < C; ++c) { float v = xr[c]; acc += v; mx = fmaxf(mx, v); }
    mm[r] = acc / C;
    mm[rows + r] = mx;
}

// 7x7 conv (2 in-ch -> 1), pad 3, sigmoid, no bias
__global__ void k_saconv(const float* __restrict__ mm, const float* __restrict__ w,
                         float* __restrict__ sa, int Bn, int Hn, int Wn)
{
    size_t idx = (size_t)blockIdx.x * blockDim.x + threadIdx.x;
    size_t rows = (size_t)Bn * Hn * Wn;
    if (idx >= rows) return;
    int wq = (int)(idx % Wn);
    size_t p = idx / Wn;
    int hq = (int)(p % Hn);
    int bb = (int)(p / Hn);
    float s = 0.f;
    for (int ci = 0; ci < 2; ++ci)
        for (int dy = 0; dy < 7; ++dy) {
            int hy = hq + dy - 3;
            if (hy < 0 || hy >= Hn) continue;
            for (int dx = 0; dx < 7; ++dx) {
                int wx = wq + dx - 3;
                if (wx < 0 || wx >= Wn) continue;
                s += mm[(size_t)ci * rows + ((size_t)bb * Hn + hy) * Wn + wx]
                     * w[ci * 49 + dy * 7 + dx];
            }
        }
    sa[idx] = sigmoid_f(s);
}

__global__ void k_nchw2nhwc(const float* __restrict__ src, float* __restrict__ dst,
                            int Bn, int Cn, int Hn, int Wn)
{
    size_t idx = (size_t)blockIdx.x * blockDim.x + threadIdx.x;
    size_t total = (size_t)Bn * Cn * Hn * Wn;
    if (idx >= total) return;
    int c = (int)(idx % Cn);
    size_t p = idx / Cn;
    int w = (int)(p % Wn); p /= Wn;
    int h = (int)(p % Hn); p /= Hn;
    int b = (int)p;
    dst[idx] = src[(((size_t)b * Cn + c) * Hn + h) * Wn + w];
}

// out_nchw = x_nchw + o_nhwc (transposed)
__global__ void k_final(const float* __restrict__ o, const float* __restrict__ x,
                        float* __restrict__ out, int Bn, int Cn, int Hn, int Wn)
{
    size_t idx = (size_t)blockIdx.x * blockDim.x + threadIdx.x;
    size_t total = (size_t)Bn * Cn * Hn * Wn;
    if (idx >= total) return;
    int w = (int)(idx % Wn);
    size_t p = idx / Wn;
    int h = (int)(p % Hn); p /= Hn;
    int c = (int)(p % Cn);
    int b = (int)(p / Cn);
    out[idx] = x[idx] + o[(((size_t)b * Hn + h) * Wn + w) * Cn + c];
}

// ============================================================================
// Host-side orchestration
// ============================================================================
// Param index map (insertion order of setup_inputs / _make_params):
enum {
    PI_X = 0, PI_XT_W, PI_XT_B,
    PI_DWT = 3,   // 4 lifts x {dw_w,dw_b,pw_w,pw_b}: P_horz,U_horz,P_vert,U_vert
    PI_IDWT = 19,
    PI_LN_LOW_G = 35, PI_LN_LOW_B, PI_QL_W, PI_QL_B, PI_KL_W, PI_KL_B,
    PI_LN_DL_G, PI_LN_DL_B, PI_DICT_LOW,
    PI_RW1, PI_RB1, PI_RW2, PI_RB2, PI_RW3, PI_RB3,
    PI_EBIAS, PI_EXPERTS,
    PI_LN_DH_G, PI_LN_DH_B, PI_LN_H_G, PI_LN_H_B,
    PI_QH_W, PI_QH_B, PI_KH_W, PI_KH_B, PI_VA_W, PI_VA_B,
    PI_MSA_S_W = 62, PI_MSA_S_B, PI_D1_DW_W, PI_D1_DW_B, PI_D1_PW_W, PI_D1_PW_B,
    PI_D2_DW_W, PI_D2_DW_B, PI_D2_PW_W, PI_D2_PW_B, PI_D3_W, PI_D3_B, PI_SA_W,
    PI_LN_SC_G, PI_LN_SC_B, PI_RS1, PI_LN_MLP_G, PI_LN_MLP_B,
    PI_FC1_W, PI_FC1_B, PI_MLP_DW_W, PI_MLP_DW_B, PI_FC2_W, PI_FC2_B,
    PI_RS2, PI_OUT_W, PI_OUT_B
};

extern "C" void kernel_launch(void* const* d_in, const int* in_sizes, int n_in,
                              void* d_out, int out_size, void* d_ws, size_t ws_size,
                              hipStream_t stream)
{
    (void)in_sizes; (void)n_in; (void)out_size; (void)ws_size;
    const float* P[96];
    for (int i = 0; i < 96 && i < n_in; ++i) P[i] = (const float*)d_in[i];

    const int Bn = 8, Cc = 128, Hn = 128, Wn = 128;
    const size_t rowsF = (size_t)Bn * Hn * Wn;          // 131072
    const size_t rowsH = (size_t)Bn * Hn * (Wn / 2);    // 65536
    const size_t rowsQ = (size_t)Bn * (Hn / 2) * (Wn / 2); // 32768

    float* ws = (float*)d_ws;
    size_t off = 0;
    auto alloc = [&](size_t n) { float* p = ws + off; off += n; return p; };

    // persistent across phases
    float* recon = alloc(rowsF * 128);
    const size_t mark = off;

    // ----- phase 1 buffers (through idwt) -----
    float* xnh  = alloc(rowsF * 128);
    float* emb  = alloc(rowsF * 128);
    float* evH  = alloc(rowsH * 128);
    float* odH  = alloc(rowsH * 128);
    float* hho  = alloc(rowsH * 128);   // h_horz
    float* lho  = alloc(rowsH * 128);   // l_horz
    float* lt1  = alloc(rowsH * 128);   // lifting dw temp
    float* lt2  = alloc(rowsH * 128);   // lifting pw out
    float* eQ   = alloc(rowsQ * 128);
    float* oQ   = alloc(rowsQ * 128);
    float* hll  = alloc(rowsQ * 128);
    float* llq  = alloc(rowsQ * 128);
    float* eB   = alloc(rowsQ * 128);
    float* oB   = alloc(rowsQ * 128);
    float* hhh  = alloc(rowsQ * 128);
    float* lhq  = alloc(rowsQ * 128);
    float* hf   = alloc(rowsQ * 384);
    float* lln  = alloc(rowsQ * 128);
    float* qlow = alloc(rowsQ * 128);
    float* dln  = alloc((size_t)64 * 128);
    float* kdm  = alloc((size_t)64 * 128);
    float* siml = alloc(rowsQ * 64);
    float* llp  = alloc(rowsQ * 128);
    float* cat  = alloc(rowsQ * 512);
    float* r1b  = alloc(rowsQ * 384);
    float* r2b  = alloc(rowsQ * 96);
    float* lgts = alloc(rowsQ * 4);
    float* rwb  = alloc(rowsQ * 4);
    float* hfln = alloc(rowsQ * 384);
    float* qh   = alloc(rowsQ * 384);
    float* ehln = alloc((size_t)256 * 384);
    float* keys = alloc((size_t)256 * 384);
    float* vmat = alloc((size_t)256 * 384);
    float* simh = alloc(rowsQ * 256);
    float* eo   = alloc(rowsQ * 384);
    float* hfp  = alloc(rowsQ * 384);
    float* lhc  = alloc(rowsQ * 128);
    float* hlc  = alloc(rowsQ * 128);
    float* hhc  = alloc(rowsQ * 128);
    float* evq  = alloc(rowsQ * 128);
    float* odq  = alloc(rowsQ * 128);
    float* lhz  = alloc(rowsH * 128);
    float* hhz  = alloc(rowsH * 128);
    float* evh2 = alloc(rowsH * 128);
    float* odh2 = alloc(rowsH * 128);

    auto gemm = [&](const float* Aa, const float* Wm, const float* bias, float* Co,
                    int M, int N, int K, int lda, int ldc, int transB, int epi, float alpha) {
        dim3 g((unsigned)((N + 63) / 64), (unsigned)((M + 63) / 64), 1);
        k_gemm<<<g, 128, 0, stream>>>(Aa, Wm, bias, Co, M, N, K, lda, ldc, transB, epi, alpha);
    };
    auto dwc = [&](const float* x, const float* w, const float* b, float* y,
                   int Bx, int Hx, int Wx, int Cx, int ldx, int offx, int pre, int post) {
        size_t tot = (size_t)Bx * Hx * Wx * Cx;
        k_dwconv3x3<<<(unsigned)((tot + 255) / 256), 256, 0, stream>>>(
            x, w, b, y, Bx, Hx, Wx, Cx, ldx, offx, pre, post);
    };
    auto lift = [&](const float* x, int pbase, int j, float* out, int Bx, int Hx, int Wx) {
        const float* dww = P[pbase + 4 * j + 0];
        const float* dwb = P[pbase + 4 * j + 1];
        const float* pww = P[pbase + 4 * j + 2];
        const float* pwb = P[pbase + 4 * j + 3];
        dwc(x, dww, dwb, lt1, Bx, Hx, Wx, 128, 128, 0, 0, 1);            // gelu(dw(x))
        gemm(lt1, pww, pwb, out, Bx * Hx * Wx, 128, 128, 128, 128, 1, 0, 1.0f); // pw 'oc'
    };
    auto addsub = [&](const float* a, const float* b, float* o, size_t n, float beta) {
        k_addsub<<<(unsigned)((n + 255) / 256), 256, 0, stream>>>(a, b, o, n, beta);
    };
    auto copych = [&](const float* s, float* d, size_t rows, int Cx,
                      int lds, int offs, int ldd, int offd) {
        size_t tot = rows * (size_t)Cx;
        k_copych<<<(unsigned)((tot + 255) / 256), 256, 0, stream>>>(s, d, rows, Cx, lds, offs, ldd, offd);
    };
    auto slice = [&](const float* s, float* d, int Bx, int Ho, int Wo, int Cx,
                     int Hi, int Wi, int axis, int par) {
        size_t tot = (size_t)Bx * Ho * Wo * Cx;
        k_slice<<<(unsigned)((tot + 255) / 256), 256, 0, stream>>>(s, d, Bx, Ho, Wo, Cx, Hi, Wi, axis, par);
    };
    auto interl = [&](const float* e, const float* o, float* d, int Bx, int Ho, int Wo, int Cx, int axis) {
        size_t tot = (size_t)Bx * Ho * Wo * Cx;
        k_interleave<<<(unsigned)((tot + 255) / 256), 256, 0, stream>>>(e, o, d, Bx, Ho, Wo, Cx, axis);
    };

    // ---------------- x_trans ----------------
    k_nchw2nhwc<<<(unsigned)((rowsF * 128 + 255) / 256), 256, 0, stream>>>(P[PI_X], xnh, Bn, Cc, Hn, Wn);
    gemm(xnh, P[PI_XT_W], P[PI_XT_B], emb, (int)rowsF, 128, 128, 128, 128, 0, 0, 1.0f);

    // ---------------- DWT ----------------
    slice(emb, evH, Bn, Hn, Wn / 2, 128, Hn, Wn, 3, 0);
    slice(emb, odH, Bn, Hn, Wn / 2, 128, Hn, Wn, 3, 1);
    lift(evH, PI_DWT, 0, lt2, Bn, Hn, Wn / 2);           // P_horz
    addsub(odH, lt2, hho, rowsH * 128, -1.0f);
    lift(hho, PI_DWT, 1, lt2, Bn, Hn, Wn / 2);           // U_horz
    addsub(evH, lt2, lho, rowsH * 128, 1.0f);

    slice(lho, eQ, Bn, Hn / 2, Wn / 2, 128, Hn, Wn / 2, 2, 0);
    slice(lho, oQ, Bn, Hn / 2, Wn / 2, 128, Hn, Wn / 2, 2, 1);
    lift(eQ, PI_DWT, 2, lt2, Bn, Hn / 2, Wn / 2);        // P_vert
    addsub(oQ, lt2, hll, rowsQ * 128, -1.0f);
    lift(hll, PI_DWT, 3, lt2, Bn, Hn / 2, Wn / 2);       // U_vert
    addsub(eQ, lt2, llq, rowsQ * 128, 1.0f);

    slice(hho, eB, Bn, Hn / 2, Wn / 2, 128, Hn, Wn / 2, 2, 0);
    slice(hho, oB, Bn, Hn / 2, Wn / 2, 128, Hn, Wn / 2, 2, 1);
    lift(eB, PI_DWT, 2, lt2, Bn, Hn / 2, Wn / 2);
    addsub(oB, lt2, hhh, rowsQ * 128, -1.0f);
    lift(hhh, PI_DWT, 3, lt2, Bn, Hn / 2, Wn / 2);
    addsub(eB, lt2, lhq, rowsQ * 128, 1.0f);

    copych(hll, hf, rowsQ, 128, 128, 0, 384, 0);
    copych(lhq, hf, rowsQ, 128, 128, 0, 384, 128);
    copych(hhh, hf, rowsQ, 128, 128, 0, 384, 256);

    // ---------------- low-freq dictionary attention ----------------
    k_layernorm<<<(unsigned)rowsQ, 128, 0, stream>>>(llq, P[PI_LN_LOW_G], P[PI_LN_LOW_B], lln, 128);
    gemm(lln, P[PI_QL_W], P[PI_QL_B], qlow, (int)rowsQ, 128, 128, 128, 128, 0, 0, 1.0f);
    k_layernorm<<<64, 128, 0, stream>>>(P[PI_DICT_LOW], P[PI_LN_DL_G], P[PI_LN_DL_B], dln, 128);
    gemm(dln, P[PI_KL_W], P[PI_KL_B], kdm, 64, 128, 128, 128, 128, 0, 0, 1.0f);
    gemm(qlow, kdm, nullptr, siml, (int)rowsQ, 64, 128, 128, 64, 1, 0, 0.08838834764831845f);
    k_softmax<<<(unsigned)rowsQ, 64, 0, stream>>>(siml, 64);
    gemm(siml, P[PI_DICT_LOW], nullptr, llp, (int)rowsQ, 128, 64, 64, 128, 0, 0, 1.0f);

    // ---------------- high-freq MoE attention ----------------
    copych(hf, cat, rowsQ, 384, 384, 0, 512, 0);
    copych(llp, cat, rowsQ, 128, 128, 0, 512, 384);
    gemm(cat, P[PI_RW1], P[PI_RB1], r1b, (int)rowsQ, 384, 512, 512, 384, 0, 1, 1.0f);
    gemm(r1b, P[PI_RW2], P[PI_RB2], r2b, (int)rowsQ, 96, 384, 384, 96, 0, 1, 1.0f);
    gemm(r2b, P[PI_RW3], P[PI_RB3], lgts, (int)rowsQ, 4, 96, 96, 4, 0, 0, 1.0f);
    k_router<<<(unsigned)((rowsQ + 255) / 256), 256, 0, stream>>>(lgts, P[PI_EBIAS], rwb, (int)rowsQ);

    k_layernorm<<<(unsigned)rowsQ, 128, 0, stream>>>(hf, P[PI_LN_H_G], P[PI_LN_H_B], hfln, 384);
    gemm(hfln, P[PI_QH_W], P[PI_QH_B], qh, (int)rowsQ, 384, 384, 384, 384, 0, 0, 1.0f);
    k_layernorm<<<256, 128, 0, stream>>>(P[PI_EXPERTS], P[PI_LN_DH_G], P[PI_LN_DH_B], ehln, 384);
    gemm(ehln, P[PI_KH_W], P[PI_KH_B], keys, 256, 384, 384, 384, 384, 0, 0, 1.0f);
    gemm(qh, keys, nullptr, simh, (int)rowsQ, 256, 384, 384, 256, 1, 0, 0.05103103630798288f);
    k_softmax<<<(unsigned)(rowsQ * 4), 64, 0, stream>>>(simh, 64);
    gemm(P[PI_EXPERTS], P[PI_VA_W], P[PI_VA_B], vmat, 256, 384, 384, 384, 384, 0, 0, 1.0f);

    hipMemsetAsync(hfp, 0, rowsQ * 384 * sizeof(float), stream);
    for (int k = 0; k < 4; ++k) {
        gemm(simh + k * 64, vmat + (size_t)k * 64 * 384, nullptr, eo,
             (int)rowsQ, 384, 64, 256, 384, 0, 0, 1.0f);
        k_combine<<<(unsigned)((rowsQ * 384 + 255) / 256), 256, 0, stream>>>(eo, rwb, hfp, rowsQ, k);
    }

    // ---------------- IDWT ----------------
    copych(hfp, lhc, rowsQ, 128, 384, 0, 128, 0);
    copych(hfp, hlc, rowsQ, 128, 384, 128, 128, 0);
    copych(hfp, hhc, rowsQ, 128, 384, 256, 128, 0);

    lift(lhc, PI_IDWT, 3, lt2, Bn, Hn / 2, Wn / 2);      // U_vert
    addsub(llp, lt2, evq, rowsQ * 128, -1.0f);
    lift(evq, PI_IDWT, 2, lt2, Bn, Hn / 2, Wn / 2);      // P_vert
    addsub(lhc, lt2, odq, rowsQ * 128, 1.0f);
    interl(evq, odq, lhz, Bn, Hn, Wn / 2, 128, 2);

    lift(hhc, PI_IDWT, 3, lt2, Bn, Hn / 2, Wn / 2);
    addsub(hlc, lt2, evq, rowsQ * 128, -1.0f);
    lift(evq, PI_IDWT, 2, lt2, Bn, Hn / 2, Wn / 2);
    addsub(hhc, lt2, odq, rowsQ * 128, 1.0f);
    interl(evq, odq, hhz, Bn, Hn, Wn / 2, 128, 2);

    lift(hhz, PI_IDWT, 1, lt2, Bn, Hn, Wn / 2);          // U_horz
    addsub(lhz, lt2, evh2, rowsH * 128, -1.0f);
    lift(evh2, PI_IDWT, 0, lt2, Bn, Hn, Wn / 2);         // P_horz
    addsub(hhz, lt2, odh2, rowsH * 128, 1.0f);
    interl(evh2, odh2, recon, Bn, Hn, Wn, 128, 3);

    // ----- phase 2: reuse phase-1 arena -----
    off = mark;
    float* lnb  = alloc(rowsF * 128);
    float* mA   = alloc(rowsF * 128);
    float* mB   = alloc(rowsF * 128);
    float* mC   = alloc(rowsF * 128);
    float* mm   = alloc(2 * rowsF);
    float* sab  = alloc(rowsF);
    float* fbuf = alloc(rowsF * 512);
    float* dwa  = alloc(rowsF * 256);
    float* gbuf = alloc(rowsF * 256);
    float* obuf = alloc(rowsF * 128);

    // ---------------- MSA ----------------
    k_layernorm<<<(unsigned)rowsF, 128, 0, stream>>>(recon, P[PI_LN_SC_G], P[PI_LN_SC_B], lnb, 128);
    gemm(lnb, P[PI_MSA_S_W], P[PI_MSA_S_B], mA, (int)rowsF, 128, 128, 128, 128, 1, 0, 1.0f);
    dwc(mA, P[PI_D1_DW_W], P[PI_D1_DW_B], mB, Bn, Hn, Wn, 128, 128, 0, 1, 0);
    gemm(mB, P[PI_D1_PW_W], P[PI_D1_PW_B], mC, (int)rowsF, 128, 128, 128, 128, 1, 0, 1.0f);
    dwc(mC, P[PI_D2_DW_W], P[PI_D2_DW_B], mB, Bn, Hn, Wn, 128, 128, 0, 1, 0);
    gemm(mB, P[PI_D2_PW_W], P[PI_D2_PW_B], mA, (int)rowsF, 128, 128, 128, 128, 1, 0, 1.0f);
    gemm(mA, P[PI_D3_W], P[PI_D3_B], mC, (int)rowsF, 128, 128, 128, 128, 1, 0, 1.0f);  // s_out
    k_poolmm<<<(unsigned)((rowsF + 255) / 256), 256, 0, stream>>>(mC, mm, rowsF, 128);
    k_saconv<<<(unsigned)((rowsF + 255) / 256), 256, 0, stream>>>(mm, P[PI_SA_W], sab, Bn, Hn, Wn);
    k_scale_res<<<(unsigned)((rowsF * 128 + 255) / 256), 256, 0, stream>>>(
        recon, mC, P[PI_RS1], sab, rowsF, 128);

    // ---------------- conv-GELU MLP ----------------
    k_layernorm<<<(unsigned)rowsF, 128, 0, stream>>>(recon, P[PI_LN_MLP_G], P[PI_LN_MLP_B], lnb, 128);
    gemm(lnb, P[PI_FC1_W], P[PI_FC1_B], fbuf, (int)rowsF, 512, 128, 128, 512, 0, 0, 1.0f);
    dwc(fbuf, P[PI_MLP_DW_W], P[PI_MLP_DW_B], dwa, Bn, Hn, Wn, 256, 512, 0, 0, 0); // on a-half
    k_gelu_mul<<<(unsigned)((rowsF * 256 + 255) / 256), 256, 0, stream>>>(
        dwa, fbuf, gbuf, rowsF, 256, 512, 256);
    gemm(gbuf, P[PI_FC2_W], P[PI_FC2_B], mA, (int)rowsF, 128, 256, 256, 128, 0, 0, 1.0f);
    k_scale_res<<<(unsigned)((rowsF * 128 + 255) / 256), 256, 0, stream>>>(
        recon, mA, P[PI_RS2], nullptr, rowsF, 128);

    // ---------------- output projection + residual ----------------
    gemm(recon, P[PI_OUT_W], P[PI_OUT_B], obuf, (int)rowsF, 128, 128, 128, 128, 0, 0, 1.0f);
    k_final<<<(unsigned)((rowsF * 128 + 255) / 256), 256, 0, stream>>>(
        obuf, P[PI_X], (float*)d_out, Bn, Cc, Hn, Wn);
}
